// GPT_7052336300377
// MI455X (gfx1250) — matrix operand order repaired and checked by reading, not compiled, not run on hardware
//
#include <hip/hip_runtime.h>
#include <hip/hip_bf16.h>
#include <math.h>

// ---------------- model constants ----------------
#define NL 12
#define BB 4
#define TS 1024
#define CH 768
#define NH 12
#define VO 50257
#define HD 64
#define MM (BB*TS)     // 4096 rows
#define FF (4*CH)      // 3072
#define ASCALE 0.125f  // 1/sqrt(64)

typedef __attribute__((ext_vector_type(16))) __bf16 v16bf;
typedef __attribute__((ext_vector_type(8)))  float  v8f;

__device__ __forceinline__ v8f wmma_bf16(v16bf a, v16bf b, v8f c) {
  // D = A(16x32 bf16) x B(32x16 bf16) + C(16x16 f32)
  return __builtin_amdgcn_wmma_f32_16x16x32_bf16(false, a, false, b, (short)0, c,
                                                 false, false);
}

__device__ __forceinline__ v8f zero8() {
  v8f z;
#pragma unroll
  for (int i = 0; i < 8; ++i) z[i] = 0.0f;
  return z;
}

union FragB { v16bf v; unsigned int u[8]; };
union PackB { __bf16 h[2]; unsigned int u; };

// generic-pointer -> 32-bit LDS offset (LDS aperture: addr[31:0])
__device__ __forceinline__ unsigned int lds_off(const void* p) {
  return (unsigned int)(unsigned long long)p;
}

// async global->LDS copy of 16 bytes per lane (CDNA5, ASYNCcnt-tracked)
__device__ __forceinline__ void async_copy_b128(unsigned int lds_addr,
                                                const void* gaddr) {
  asm volatile("global_load_async_to_lds_b128 %0, %1, off"
               :: "v"(lds_addr), "v"(gaddr) : "memory");
}
__device__ __forceinline__ void wait_async0() {
  asm volatile("s_wait_asynccnt 0x0" ::: "memory");
}

// ---------------- embedding: x = tok_emb[idx] + pos_emb ----------------
__launch_bounds__(256)
__global__ void embed_kernel(const int* __restrict__ idx,
                             const float* __restrict__ tok,
                             const float* __restrict__ pos,
                             float* __restrict__ x) {
  size_t i = (size_t)blockIdx.x * 256 + threadIdx.x;
  if (i >= (size_t)MM * CH) return;
  int c = (int)(i % CH);
  size_t bt = i / CH;
  int t = (int)(bt % TS);
  int token = idx[bt];
  x[i] = tok[(size_t)token * CH + c] + pos[(size_t)t * CH + c];
}

// ---------------- layernorm -> bf16 ----------------
__launch_bounds__(256)
__global__ void layernorm_bf16(const float* __restrict__ x,
                               const float* __restrict__ g,
                               const float* __restrict__ be,
                               __bf16* __restrict__ out) {
  __shared__ float red[256];
  int row = blockIdx.x;
  int tid = threadIdx.x;
  const float* xr = x + (size_t)row * CH;

  float xl[3];
#pragma unroll
  for (int ii = 0; ii < 3; ++ii) xl[ii] = xr[tid + ii * 256];
  float s = xl[0] + xl[1] + xl[2];
  red[tid] = s; __syncthreads();
  for (int st = 128; st > 0; st >>= 1) {
    if (tid < st) red[tid] += red[tid + st];
    __syncthreads();
  }
  float mean = red[0] * (1.0f / CH);
  __syncthreads();

  float s2 = 0.f;
#pragma unroll
  for (int ii = 0; ii < 3; ++ii) { float d = xl[ii] - mean; s2 += d * d; }
  red[tid] = s2; __syncthreads();
  for (int st = 128; st > 0; st >>= 1) {
    if (tid < st) red[tid] += red[tid + st];
    __syncthreads();
  }
  float rstd = rsqrtf(red[0] * (1.0f / CH) + 1e-5f);

  __bf16* orow = out + (size_t)row * CH;
#pragma unroll
  for (int ii = 0; ii < 3; ++ii) {
    int c = tid + ii * 256;
    orow[c] = (__bf16)((xl[ii] - mean) * rstd * g[c] + be[c]);
  }
}

// ---------------- WMMA GEMM: out[M,N] = A_bf16[M,K] @ W_f32[K,N] (+bias,+gelu,+res) ----
#define TM 128
#define TN 128
#define TK 32

__launch_bounds__(256)
__global__ void gemm_bf16_wmma(const __bf16* __restrict__ A,
                               const float* __restrict__ W,
                               const float* __restrict__ bias,
                               const float* __restrict__ residual,
                               float* __restrict__ outF,
                               __bf16* __restrict__ outB,
                               int M, int N, int K, int gelu, int vstore) {
  __shared__ __bf16 sA[2][TM][TK];     // row-major [m][k], 2 x 8 KB (async-filled)
  __shared__ __bf16 sBt[2][TN][TK];    // transposed [n][k], 2 x 8 KB

  int tid = threadIdx.x;
  int wave = tid >> 5, lane = tid & 31;
  int hl = lane >> 4, l16 = lane & 15;
  int wm = wave >> 2, wn = wave & 3;    // 2x4 wave grid; each wave: 64x32 output
  int m0 = blockIdx.y * TM;
  int n0 = blockIdx.x * TN;

  // A staging: 128x32 bf16 = 8 KB = 256 lanes x 2 x b128
  int aRow0 = tid >> 2;                 // 0..63
  int aByte = (tid & 3) * 16;
  unsigned int sAbase = lds_off(&sA[0][0][0]);
  // B staging: 32x128 f32 -> bf16; each lane: fixed column, 8 k-adjacent pairs
  int bN = tid & 127;
  int bP = tid >> 7;                    // 0..1
  bool inb = (n0 + bN) < N;             // per-lane (head edge tile only)
  bool fullN = (n0 + TN) <= N;          // block-uniform

  int nIter = K / TK;

  v8f acc[4][2];
#pragma unroll
  for (int i = 0; i < 4; ++i) { acc[i][0] = zero8(); acc[i][1] = zero8(); }

  auto stage = [&](int it) {
    int k0 = it * TK;
    int buf = it & 1;
    // async A (bf16 already in memory)
    const char* ga0 = (const char*)(A + (size_t)(m0 + aRow0) * K + k0) + aByte;
    const char* ga1 = ga0 + (size_t)64 * K * sizeof(__bf16);
    async_copy_b128(sAbase + buf * 8192 + tid * 16, ga0);
    async_copy_b128(sAbase + buf * 8192 + 4096 + tid * 16, ga1);
    // W tile: batched f32 loads, pair-packed bf16 stores
    const float* wbase = W + (size_t)k0 * N + n0 + bN;
    if (it + 1 < nIter)
      __builtin_prefetch(wbase + (size_t)TK * N, 0, 1);
    float la[8], lb[8];
    if (fullN) {
#pragma unroll
      for (int ii = 0; ii < 8; ++ii) {
        int kA = 2 * (bP + 2 * ii);
        la[ii] = wbase[(size_t)kA * N];
        lb[ii] = wbase[(size_t)(kA + 1) * N];
      }
    } else {
#pragma unroll
      for (int ii = 0; ii < 8; ++ii) {
        int kA = 2 * (bP + 2 * ii);
        la[ii] = inb ? wbase[(size_t)kA * N] : 0.f;
        lb[ii] = inb ? wbase[(size_t)(kA + 1) * N] : 0.f;
      }
    }
#pragma unroll
    for (int ii = 0; ii < 8; ++ii) {
      int kA = 2 * (bP + 2 * ii);
      PackB pk;
      pk.h[0] = (__bf16)la[ii];
      pk.h[1] = (__bf16)lb[ii];
      *(unsigned int*)&sBt[buf][bN][kA] = pk.u;
    }
  };

  stage(0);
  for (int it = 0; it < nIter; ++it) {
    int buf = it & 1;
    wait_async0();
    __syncthreads();
    if (it + 1 < nIter) stage(it + 1);   // overlap next-tile staging with compute

    FragB bF[2];
#pragma unroll
    for (int t = 0; t < 2; ++t) {
      int nB = t * 64 + wn * 16 + l16;
#pragma unroll
      for (int j = 0; j < 8; ++j) {
        int kk = hl * 16 + 2 * j;
        bF[t].u[j] = *((const unsigned int*)&sBt[buf][nB][0] + (kk >> 1));
      }
    }
#pragma unroll
    for (int sub = 0; sub < 4; ++sub) {
      FragB aF;
      int mA = wm * 64 + sub * 16 + l16;
#pragma unroll
      for (int i = 0; i < 8; ++i) {
        int kk = hl * 16 + 2 * i;
        aF.u[i] = *((const unsigned int*)&sA[buf][mA][0] + (kk >> 1));
      }
      acc[sub][0] = wmma_bf16(aF.v, bF[0].v, acc[sub][0]);
      acc[sub][1] = wmma_bf16(aF.v, bF[1].v, acc[sub][1]);
    }
  }

  // ---- epilogue ----
#pragma unroll
  for (int t = 0; t < 2; ++t) {
    int n = n0 + t * 64 + wn * 16 + l16;
    if (n >= N) continue;
    float bv = bias ? bias[n] : 0.f;
#pragma unroll
    for (int sub = 0; sub < 4; ++sub) {
#pragma unroll
      for (int r = 0; r < 8; ++r) {
        int m = m0 + wm * 64 + sub * 16 + r + 8 * hl;
        float vv = acc[sub][t][r] + bv;
        if (gelu) vv = 0.5f * vv * (1.0f + erff(vv * 0.70710678118654752f));
        if (residual) vv += residual[(size_t)m * N + n];
        if (outF) outF[(size_t)m * N + n] = vv;
        if (outB) {
          if (vstore) {  // store V transposed: [B,H,D,T]
            int hh = n >> 6, dd = n & (HD - 1);
            int bb = m >> 10, tt = m & (TS - 1);
            outB[(((size_t)bb * NH + hh) * HD + dd) * TS + tt] = (__bf16)vv;
          } else {
            outB[(size_t)m * N + n] = (__bf16)vv;
          }
        }
      }
    }
  }
}

// ---------------- flash attention (wave-autonomous, online softmax) ----------------
// q,k: bf16 [B,T,C]   vt: bf16 [B,H,D,T]   y: bf16 [B,T,C]
__launch_bounds__(256)
__global__ void attn_flash(const __bf16* __restrict__ q,
                           const __bf16* __restrict__ kq,
                           const __bf16* __restrict__ vt,
                           __bf16* __restrict__ y) {
  __shared__ __bf16 sP[8][16][32];  // per-wave P tile (16 rows x 32 keys)
  int tid = threadIdx.x;
  int wave = tid >> 5, lane = tid & 31;
  int hl = lane >> 4, l16 = lane & 15;
  int b = blockIdx.z, h = blockIdx.y;
  int q0 = blockIdx.x * 128 + wave * 16;  // 16 query rows per wave

  // Q A-fragments (two K=32 chains covering d=0..63), loaded once
  const __bf16* qp = q + ((size_t)(b * TS + q0 + l16) * CH + h * HD);
  FragB qa[2];
#pragma unroll
  for (int c = 0; c < 2; ++c)
#pragma unroll
    for (int i = 0; i < 8; ++i) {
      int d = c * 32 + hl * 16 + 2 * i;
      qa[c].u[i] = *(const unsigned int*)(qp + d);
    }

  v8f acc[4];
#pragma unroll
  for (int i = 0; i < 4; ++i) acc[i] = zero8();
  float mst[8], lst[8];
#pragma unroll
  for (int r = 0; r < 8; ++r) { mst[r] = -1e30f; lst[r] = 0.f; }

  int nkt = (q0 + 15) / 32 + 1;  // causal: key tiles of 32
  for (int kt = 0; kt < nkt; ++kt) {
    int kbase = kt * 32;
    v8f s[2];
#pragma unroll
    for (int nt = 0; nt < 2; ++nt) {
      int key = kbase + nt * 16 + l16;
      const __bf16* kp = kq + ((size_t)(b * TS + key) * CH + h * HD);
      FragB kb0, kb1;
#pragma unroll
      for (int j = 0; j < 8; ++j) {
        int d = hl * 16 + 2 * j;
        kb0.u[j] = *(const unsigned int*)(kp + d);
        kb1.u[j] = *(const unsigned int*)(kp + d + 32);
      }
      v8f z = zero8();
      z = wmma_bf16(qa[0].v, kb0.v, z);
      z = wmma_bf16(qa[1].v, kb1.v, z);
      s[nt] = z;
    }
    // scale + causal mask + online softmax (rows live at r+8*hl, cols at lane%16)
#pragma unroll
    for (int r = 0; r < 8; ++r) {
      int m = q0 + r + 8 * hl;
      float v0 = s[0][r] * ASCALE; if (kbase + l16 > m)      v0 = -1e30f;
      float v1 = s[1][r] * ASCALE; if (kbase + 16 + l16 > m) v1 = -1e30f;
      float mr = fmaxf(v0, v1);
#pragma unroll
      for (int msk = 1; msk < 16; msk <<= 1) mr = fmaxf(mr, __shfl_xor(mr, msk, 32));
      float mn = fmaxf(mst[r], mr);
      float corr = __expf(mst[r] - mn);
      float p0 = __expf(v0 - mn), p1 = __expf(v1 - mn);
      float ps = p0 + p1;
#pragma unroll
      for (int msk = 1; msk < 16; msk <<= 1) ps += __shfl_xor(ps, msk, 32);
      lst[r] = lst[r] * corr + ps;
      mst[r] = mn;
      acc[0][r] *= corr; acc[1][r] *= corr; acc[2][r] *= corr; acc[3][r] *= corr;
      sP[wave][r + 8 * hl][l16]      = (__bf16)p0;
      sP[wave][r + 8 * hl][16 + l16] = (__bf16)p1;
    }
    // re-swizzle P from C-frag layout (via this wave's LDS tile) to A-frag layout
    FragB pF;
#pragma unroll
    for (int i = 0; i < 8; ++i) {
      int kk = hl * 16 + 2 * i;
      pF.u[i] = *((const unsigned int*)&sP[wave][l16][0] + (kk >> 1));
    }
    // accumulate Y += P @ V  (V transposed layout -> contiguous key pairs)
#pragma unroll
    for (int dt = 0; dt < 4; ++dt) {
      int d = dt * 16 + l16;
      const __bf16* vp = vt + (((size_t)(b * NH + h) * HD + d) * TS + kbase);
      FragB vF;
#pragma unroll
      for (int j = 0; j < 8; ++j)
        vF.u[j] = *(const unsigned int*)(vp + hl * 16 + 2 * j);
      acc[dt] = wmma_bf16(pF.v, vF.v, acc[dt]);
    }
  }

  // normalize and store y (bf16, standard [B,T,C] layout)
#pragma unroll
  for (int dt = 0; dt < 4; ++dt)
#pragma unroll
    for (int r = 0; r < 8; ++r) {
      int m = q0 + r + 8 * hl;
      float ov = acc[dt][r] / lst[r];
      y[(size_t)(b * TS + m) * CH + h * HD + dt * 16 + l16] = (__bf16)ov;
    }
}

// ---------------- host-side orchestration ----------------
extern "C" void kernel_launch(void* const* d_in, const int* in_sizes, int n_in,
                              void* d_out, int out_size, void* d_ws, size_t ws_size,
                              hipStream_t stream) {
  (void)in_sizes; (void)n_in; (void)out_size; (void)ws_size;
  const int*   idx    = (const int*)  d_in[0];
  const float* tok    = (const float*)d_in[1];
  const float* pos    = (const float*)d_in[2];
  const float* wq     = (const float*)d_in[3];
  const float* bq     = (const float*)d_in[4];
  const float* wk     = (const float*)d_in[5];
  const float* bk     = (const float*)d_in[6];
  const float* wv     = (const float*)d_in[7];
  const float* bv     = (const float*)d_in[8];
  const float* wo     = (const float*)d_in[9];
  const float* bo     = (const float*)d_in[10];
  const float* ln1g   = (const float*)d_in[11];
  const float* ln1b   = (const float*)d_in[12];
  const float* ln2g   = (const float*)d_in[13];
  const float* ln2b   = (const float*)d_in[14];
  const float* w1     = (const float*)d_in[15];
  const float* b1     = (const float*)d_in[16];
  const float* w2     = (const float*)d_in[17];
  const float* b2     = (const float*)d_in[18];
  const float* lnfg   = (const float*)d_in[19];
  const float* lnfb   = (const float*)d_in[20];
  const float* headw  = (const float*)d_in[21];
  float* out = (float*)d_out;

  char* wsb = (char*)d_ws;
  size_t off = 0;
  auto alloc = [&](size_t bytes) -> void* {
    void* p = wsb + off;
    off = (off + bytes + 255) & ~(size_t)255;
    return p;
  };
  float*  x   = (float*) alloc((size_t)MM * CH * 4);
  __bf16* hbf = (__bf16*)alloc((size_t)MM * CH * 2);
  __bf16* qb  = (__bf16*)alloc((size_t)MM * CH * 2);
  __bf16* kb  = (__bf16*)alloc((size_t)MM * CH * 2);
  __bf16* vtb = (__bf16*)alloc((size_t)MM * CH * 2);
  __bf16* yb  = (__bf16*)alloc((size_t)MM * CH * 2);
  __bf16* ub  = (__bf16*)alloc((size_t)MM * FF * 2);

  dim3 blk(256);
  dim3 gEmb(((size_t)MM * CH + 255) / 256);
  dim3 gLN(MM);
  dim3 gGemmC(CH / TN, MM / TM);        // N=768
  dim3 gGemmF(FF / TN, MM / TM);        // N=3072
  dim3 gGemmV((VO + TN - 1) / TN, MM / TM);
  dim3 gAttn(TS / 128, NH, BB);

  embed_kernel<<<gEmb, blk, 0, stream>>>(idx, tok, pos, x);

  for (int l = 0; l < NL; ++l) {
    const float* wq_l = wq + (size_t)l * CH * CH;
    const float* wk_l = wk + (size_t)l * CH * CH;
    const float* wv_l = wv + (size_t)l * CH * CH;
    const float* wo_l = wo + (size_t)l * CH * CH;
    const float* w1_l = w1 + (size_t)l * CH * FF;
    const float* w2_l = w2 + (size_t)l * FF * CH;

    layernorm_bf16<<<gLN, blk, 0, stream>>>(x, ln1g + l * CH, ln1b + l * CH, hbf);

    gemm_bf16_wmma<<<gGemmC, blk, 0, stream>>>(hbf, wq_l, bq + l * CH, nullptr,
                                               nullptr, qb, MM, CH, CH, 0, 0);
    gemm_bf16_wmma<<<gGemmC, blk, 0, stream>>>(hbf, wk_l, bk + l * CH, nullptr,
                                               nullptr, kb, MM, CH, CH, 0, 0);
    gemm_bf16_wmma<<<gGemmC, blk, 0, stream>>>(hbf, wv_l, bv + l * CH, nullptr,
                                               nullptr, vtb, MM, CH, CH, 0, 1);

    attn_flash<<<gAttn, blk, 0, stream>>>(qb, kb, vtb, yb);

    // x += y @ wo + bo   (in-place residual)
    gemm_bf16_wmma<<<gGemmC, blk, 0, stream>>>(yb, wo_l, bo + l * CH, x,
                                               x, nullptr, MM, CH, CH, 0, 0);

    layernorm_bf16<<<gLN, blk, 0, stream>>>(x, ln2g + l * CH, ln2b + l * CH, hbf);

    // u = gelu(h @ w1 + b1)  (bf16 for next GEMM)
    gemm_bf16_wmma<<<gGemmF, blk, 0, stream>>>(hbf, w1_l, b1 + l * FF, nullptr,
                                               nullptr, ub, MM, FF, CH, 1, 0);
    // x += u @ w2 + b2
    gemm_bf16_wmma<<<gGemmC, blk, 0, stream>>>(ub, w2_l, b2 + l * CH, x,
                                               x, nullptr, MM, CH, FF, 0, 0);
  }

  layernorm_bf16<<<gLN, blk, 0, stream>>>(x, lnfg, lnfb, hbf);
  gemm_bf16_wmma<<<gGemmV, blk, 0, stream>>>(hbf, headw, nullptr, nullptr,
                                             out, nullptr, MM, VO, CH, 0, 0);
}
